// CycleClassificationModel_30726196035901
// MI455X (gfx1250) — matrix operand order repaired and checked by reading, not compiled, not run on hardware
//
#include <hip/hip_runtime.h>

// ---------------- problem constants (fixed by the reference) ----------------
#define HID    100
#define GATES  4
#define NT     7      // 16-wide column tiles per gate: 7*16 = 112 >= 100
#define KC     4      // K chunks of 32: 4*32 = 128 (100 used + xproj rows)
#define KDIM   (KC * 32)
#define XK0    112    // K index carrying x[...,0]
#define XK1    113    // K index carrying x[...,1]
#define ONEK   114    // K index carrying constant 1.0 (bias row)
#define WAVES  8      // waves per workgroup
#define THREADS (WAVES * 32)
#define OUTD   5

typedef __bf16 bf16;
typedef __attribute__((ext_vector_type(16))) __bf16 v16bf;
typedef __attribute__((ext_vector_type(8)))  __bf16 v8bf;
typedef __attribute__((ext_vector_type(8)))  float  v8f;

__device__ __forceinline__ float sigmoid_f(float x) {
    return 1.0f / (1.0f + __expf(-x));
}
__device__ __forceinline__ float tanh_f(float x) {
#if __has_builtin(__builtin_amdgcn_tanhf)
    return __builtin_amdgcn_tanhf(x);     // v_tanh_f32 (CDNA5 TRANS op)
#else
    return 2.0f / (1.0f + __expf(-2.0f * x)) - 1.0f;
#endif
}

// Optional input pre-transpose: X[B][T][2] -> XT[T][B][2] so the per-step
// 16-row fetch inside the LSTM loop is coalesced. One-time 98 MB pass.
__global__ void transpose_x_kernel(const float2* __restrict__ X,
                                   float2* __restrict__ XT, int Bn, int T) {
    size_t idx = (size_t)blockIdx.x * blockDim.x + threadIdx.x;
    size_t total = (size_t)Bn * T;
    if (idx >= total) return;
    int    b = (int)(idx % Bn);
    size_t t = idx / Bn;
    XT[t * Bn + b] = X[(size_t)b * T + t];   // writes coalesced over b
}

__global__ __launch_bounds__(THREADS, 1)
void lstm_wmma_kernel(const float* __restrict__ X,      // [B][T][2]
                      const float* __restrict__ XT,     // [T][B][2] (optional)
                      int use_xt,
                      const float* __restrict__ w_ih,   // [4H][2]
                      const float* __restrict__ w_hh,   // [4H][H]
                      const float* __restrict__ b_ih,   // [4H]
                      const float* __restrict__ b_hh,   // [4H]
                      const float* __restrict__ w_lin,  // [5][H+1]
                      const float* __restrict__ b_lin,  // [5]
                      float* __restrict__ out,          // [B][5]
                      int Bn, int T) {
    // ---- LDS: recurrent+input weights in WMMA-B layout + per-wave [h|x|1] ----
    __shared__ v16bf Blds[KC][GATES * NT][32];      // 112 KB
    __shared__ bf16  hbuf[WAVES][16][KDIM];         //  32 KB

    const int tid = threadIdx.x;

    // Stage the fused weight matrix as bf16 B fragments.
    // B (32x16, 16-bit) layout: lane holds column N = lane%16;
    // lanes 0-15 hold K=0..7,16..23, lanes 16-31 hold K=8..15,24..31 of the chunk.
    // K rows 0..99   : w_hh (recurrent)
    // K row  112/113 : w_ih columns 0/1   (input projection)
    // K row  114     : b_ih + b_hh        (bias, multiplied by constant 1)
    for (int idx = tid; idx < KC * GATES * NT * 32 * 16; idx += THREADS) {
        int e    = idx & 15;
        int lane = (idx >> 4) & 31;
        int tile = (idx >> 9) % (GATES * NT);
        int k    = idx / (512 * GATES * NT);
        int gate = tile / NT;
        int unit = (tile % NT) * 16 + (lane & 15);
        int kk   = k * 32 + ((lane & 16) ? 8 : 0) + e + ((e >= 8) ? 8 : 0);
        float v  = 0.0f;
        if (unit < HID) {
            int r = gate * HID + unit;
            if      (kk < HID)   v = w_hh[r * HID + kk];
            else if (kk == XK0)  v = w_ih[r * 2 + 0];
            else if (kk == XK1)  v = w_ih[r * 2 + 1];
            else if (kk == ONEK) v = b_ih[r] + b_hh[r];
        }
        ((bf16*)&Blds[k][tile][lane])[e] = (bf16)v;
    }
    // h0 = 0; K padding = 0; constant-1 row for the bias.
    for (int i = tid; i < WAVES * 16 * KDIM; i += THREADS)
        ((bf16*)hbuf)[i] = (bf16)((i % KDIM) == ONEK ? 1.0f : 0.0f);
    __syncthreads();

    const int wave    = tid >> 5;
    const int lane    = tid & 31;
    const int rowbase = blockIdx.x * (WAVES * 16) + wave * 16;
    const int lrow    = lane & 15;                    // A-row / C-column index
    const int hi      = (lane & 16) ? 1 : 0;
    const int abase   = hi ? 8 : 0;                   // 16-bit A layout K base
    const v8f  zacc   = {};

    float c[NT][8];                                   // cell state in VGPRs
#pragma unroll
    for (int j = 0; j < NT; ++j)
#pragma unroll
        for (int r = 0; r < 8; ++r) c[j][r] = 0.0f;

    for (int t = 0; t < T; ++t) {
        // ---- x[rowbase+row, t, comp] -> hbuf[row][XK0+comp] (1 load/lane) ----
        {
            int xr = lane >> 1, comp = lane & 1;
            float xv = use_xt
                ? XT[((size_t)t * Bn + rowbase + xr) * 2 + comp]
                : X[((size_t)(rowbase + xr) * T + t) * 2 + comp];
            hbuf[wave][xr][XK0 + comp] = (bf16)xv;    // same-wave DS: in-order
        }

        // ---- rebuild A fragments (16-bit A 16x32 layout) from [h|x|1] ----
        v16bf A[KC];
        {
            const bf16* hrow = &hbuf[wave][lrow][0];
#pragma unroll
            for (int k = 0; k < KC; ++k) {
                v8bf lo = *(const v8bf*)&hrow[k * 32 + abase];
                v8bf hh = *(const v8bf*)&hrow[k * 32 + abase + 16];
                A[k] = __builtin_shufflevector(lo, hh,
                        0,1,2,3,4,5,6,7,8,9,10,11,12,13,14,15);
            }
        }

        // ---- gates = [h|x|1] @ [w_hh|w_ih|b]^T : 4 gate tiles per j ----
#pragma unroll
        for (int j = 0; j < NT; ++j) {
            // Opaque copy of the tile index: the B addresses below are no
            // longer provably loop-invariant, so the ds_load_b128s stay in
            // the loop instead of being hoisted into 896 VGPRs and spilled.
            int jj = j;
            asm volatile("" : "+v"(jj));

            v8f acc[GATES];
#pragma unroll
            for (int k = 0; k < KC; ++k) {
#pragma unroll
                for (int g = 0; g < GATES; ++g) {
                    v16bf bf = Blds[k][g * NT + jj][lane];
                    acc[g] = __builtin_amdgcn_wmma_f32_16x16x32_bf16(
                        false, A[k], false, bf, (short)0,
                        (k == 0) ? zacc : acc[g], false, false);
                }
            }

            // ---- elementwise LSTM update; i/f/g/o share (lane, r) ----
            int unit = j * 16 + lrow;
#pragma unroll
            for (int r = 0; r < 8; ++r) {
                float ig = sigmoid_f(acc[0][r]);
                float fg = sigmoid_f(acc[1][r]);
                float gg = tanh_f(acc[2][r]);
                float og = sigmoid_f(acc[3][r]);
                float cn = fg * c[j][r] + ig * gg;
                c[j][r]  = cn;
                float hn = og * tanh_f(cn);
                hbuf[wave][r + 8 * hi][unit] = (bf16)((unit < HID) ? hn : 0.0f);
            }
        }
    }

    // ---- final linear: out = [h_T, T] @ w_lin^T + b_lin ----
    if (lane < 16) {
        int row = rowbase + lane;
        for (int o = 0; o < OUTD; ++o) {
            float s = b_lin[o] + (float)T * w_lin[o * (HID + 1) + HID];
            for (int jj = 0; jj < HID; ++jj)
                s += (float)hbuf[wave][lane][jj] * w_lin[o * (HID + 1) + jj];
            out[(size_t)row * OUTD + o] = s;
        }
    }
}

extern "C" void kernel_launch(void* const* d_in, const int* in_sizes, int n_in,
                              void* d_out, int out_size, void* d_ws, size_t ws_size,
                              hipStream_t stream) {
    const float* x     = (const float*)d_in[0];
    const float* w_ih  = (const float*)d_in[1];
    const float* w_hh  = (const float*)d_in[2];
    const float* b_ih  = (const float*)d_in[3];
    const float* b_hh  = (const float*)d_in[4];
    const float* w_lin = (const float*)d_in[5];
    const float* b_lin = (const float*)d_in[6];
    float*       out   = (float*)d_out;

    const int Bn = 4096, T = 3000;

    size_t need   = (size_t)Bn * T * 2 * sizeof(float);
    int    use_xt = (d_ws != nullptr && ws_size >= need) ? 1 : 0;
    float* xt     = (float*)d_ws;

    if (use_xt) {
        size_t total  = (size_t)Bn * T;
        int    blocks = (int)((total + 255) / 256);
        transpose_x_kernel<<<blocks, 256, 0, stream>>>(
            (const float2*)x, (float2*)xt, Bn, T);
    }

    int grid = Bn / (WAVES * 16);   // 4096 / 128 = 32 workgroups
    lstm_wmma_kernel<<<grid, THREADS, 0, stream>>>(
        x, xt, use_xt, w_ih, w_hh, b_ih, b_hh, w_lin, b_lin, out, Bn, T);
}
